// ImprovedTransformerBlock_64622077936085
// MI455X (gfx1250) — matrix operand order repaired and checked by reading, not compiled, not run on hardware
//
#include <hip/hip_runtime.h>
#include <cstddef>

typedef unsigned short bf16_t;
typedef __attribute__((ext_vector_type(4)))  unsigned  uv4;
typedef __attribute__((ext_vector_type(16))) __bf16    v16bf;
typedef __attribute__((ext_vector_type(8)))  float     v8f;

union FragBF { uv4 q[2]; v16bf v; };

__device__ __forceinline__ bf16_t f2bf(float f) {
    unsigned u = __float_as_uint(f);
    unsigned r = u + 0x7FFFu + ((u >> 16) & 1u);   // round-to-nearest-even
    return (bf16_t)(r >> 16);
}
__device__ __forceinline__ float bf2f(bf16_t h) {
    return __uint_as_float(((unsigned)h) << 16);
}
__device__ __forceinline__ float gelu_f(float x) {
    return 0.5f * x * (1.0f + erff(x * 0.70710678118654752440f));
}
__device__ __forceinline__ v8f wmma_bf16(v16bf a, v16bf b, v8f c) {
    return __builtin_amdgcn_wmma_f32_16x16x32_bf16(false, a, false, b, (short)0, c, false, false);
}
// CDNA5 async memory->LDS copy (16B per lane), tracked by ASYNCcnt.
__device__ __forceinline__ void async_b128(const void* gaddr, void* lds) {
    unsigned lo = (unsigned)(unsigned long long)lds;        // low 32b = LDS byte addr
    asm volatile("global_load_async_to_lds_b128 %0, %1, off"
                 :: "v"(lo), "v"((unsigned long long)gaddr) : "memory");
}
__device__ __forceinline__ void wait_async0() {
    asm volatile("s_wait_asynccnt 0x0" ::: "memory");
}

// ---------------------------------------------------------------------------
// Generic bf16 WMMA GEMM:  C[m,n] = sum_k A[m,k] * BT[n,k]  (BT = B transposed)
// Epilogue: +bias[n] -> gelu -> *rscale[row] -> +Cin -> +resid -> store f32/bf16
// row_shift/Lb implement the dilated-conv shifted A reads with zero padding.
// Tiles are staged with GLOBAL_LOAD_ASYNC_TO_LDS_B128 (ASYNCcnt).
// ---------------------------------------------------------------------------
__global__ __launch_bounds__(256, 2)
void k_gemm(const bf16_t* __restrict__ A, const bf16_t* __restrict__ BT,
            const float* __restrict__ bias, const float* __restrict__ Cin,
            const float* __restrict__ resid, const float* __restrict__ rscale,
            int rsStride, float* __restrict__ Cf, bf16_t* __restrict__ Cb,
            int N, int K, int Lb, int shift, int doGelu)
{
    __shared__ uv4 sA4[128 * 72 / 8];
    __shared__ uv4 sB4[128 * 72 / 8];
    bf16_t* sA = (bf16_t*)sA4;
    bf16_t* sB = (bf16_t*)sB4;

    const int tid  = threadIdx.x;
    const int lane = tid & 31;
    const int wv   = tid >> 5;
    const int wm   = wv >> 1;      // 0..3 (M strips of 32)
    const int wn   = wv & 1;       // 0..1 (N strips of 64)
    const int m0   = blockIdx.y * 128;
    const int n0   = blockIdx.x * 128;

    v8f acc[2][4];
#pragma unroll
    for (int i = 0; i < 2; i++)
#pragma unroll
        for (int j = 0; j < 4; j++)
#pragma unroll
            for (int e = 0; e < 8; e++) acc[i][j][e] = 0.0f;

    for (int kt = 0; kt < K; kt += 64) {
        // ---- stage A tile async (with optional row shift + zero padding) ----
#pragma unroll
        for (int i = 0; i < 2; i++) {
            int u = tid + i * 256;          // 0..511
            int row = u >> 2, ch = u & 3;   // 4 x 16-byte chunks per 64-half row
            int rg  = m0 + row;
            int li  = (rg % Lb) + shift;
            if (li >= 0 && li < Lb) {
                async_b128(A + (size_t)(rg + shift) * K + kt + ch * 16,
                           sA + row * 72 + ch * 16);
            } else {
                uv4 z = {0, 0, 0, 0};
                *(uv4*)(sA + row * 72 + ch * 16) = z;
            }
        }
        // ---- stage B tile async ----
#pragma unroll
        for (int i = 0; i < 2; i++) {
            int u = tid + i * 256;
            int row = u >> 2, ch = u & 3;
            async_b128(BT + (size_t)(n0 + row) * K + kt + ch * 16,
                       sB + row * 72 + ch * 16);
        }
        if (kt + 64 < K) {  // speculative prefetch of next K tile
            __builtin_prefetch(A  + (size_t)(m0 + (tid >> 1 & 127)) * K + kt + 64);
            __builtin_prefetch(BT + (size_t)(n0 + (tid >> 1 & 127)) * K + kt + 64);
        }
        wait_async0();
        __syncthreads();

#pragma unroll
        for (int kk = 0; kk < 2; kk++) {
            FragBF fa[2], fb[4];
            const int offA = kk * 32 + ((lane & 16) ? 8 : 0);   // A interleaved layout
            const int offB = kk * 32 + ((lane & 16) ? 8 : 0);   // B interleaved layout
#pragma unroll
            for (int i = 0; i < 2; i++) {
                const bf16_t* p = sA + (wm * 32 + i * 16 + (lane & 15)) * 72 + offA;
                fa[i].q[0] = *(const uv4*)p;
                fa[i].q[1] = *(const uv4*)(p + 16);
            }
#pragma unroll
            for (int j = 0; j < 4; j++) {
                const bf16_t* p = sB + (wn * 64 + j * 16 + (lane & 15)) * 72 + offB;
                fb[j].q[0] = *(const uv4*)p;
                fb[j].q[1] = *(const uv4*)(p + 16);
            }
#pragma unroll
            for (int i = 0; i < 2; i++)
#pragma unroll
                for (int j = 0; j < 4; j++)
                    acc[i][j] = wmma_bf16(fa[i].v, fb[j].v, acc[i][j]);
        }
        __syncthreads();
    }

    // ---- epilogue ----
#pragma unroll
    for (int i = 0; i < 2; i++) {
        int mBase = m0 + wm * 32 + i * 16 + ((lane & 16) ? 8 : 0);
#pragma unroll
        for (int j = 0; j < 4; j++) {
            int n = n0 + wn * 64 + j * 16 + (lane & 15);
            float bv = bias ? bias[n] : 0.0f;
#pragma unroll
            for (int r = 0; r < 8; r++) {
                int row = mBase + r;
                float t = acc[i][j][r] + bv;
                if (doGelu) t = gelu_f(t);
                if (rscale) t *= rscale[(size_t)row * rsStride];
                if (Cin)    t += Cin[(size_t)row * N + n];
                if (resid)  t += resid[(size_t)row * N + n];
                if (Cf) Cf[(size_t)row * N + n] = t;
                if (Cb) Cb[(size_t)row * N + n] = f2bf(t);
            }
        }
    }
}

// ---------------------------------------------------------------------------
// Windowed multi-head attention (bidirectional), WINDOW=256, HEAD_DIM=64.
// Block = 128 threads (4 waves); each wave owns 16 query rows; block covers 64.
// grid = (4 qblocks, 16 heads, B*nw=32 windows). qkv is bf16 [8192][3072].
// ---------------------------------------------------------------------------
__global__ __launch_bounds__(128)
void k_attn(const bf16_t* __restrict__ qkv, bf16_t* __restrict__ attnO)
{
    extern __shared__ uv4 smem4[];
    bf16_t* sKV = (bf16_t*)smem4;        // K as [256][72], later V^T as [64][264]
    bf16_t* sP  = sKV + 256 * 72;        // P: 4 waves x [16][264]

    const int tid  = threadIdx.x;
    const int lane = tid & 31;
    const int wv   = tid >> 5;
    const int qblk = blockIdx.x;
    const int h    = blockIdx.y;
    const size_t rowBase = (size_t)blockIdx.z * 256;

    // ---- 1. K tile -> LDS [kidx][d] via async copy ----
#pragma unroll
    for (int it = 0; it < 16; it++) {
        int u = tid + it * 128;            // 0..2047 (256 rows x 8 chunks)
        int kidx = u >> 3, ch = u & 7;
        async_b128(qkv + (rowBase + kidx) * 3072 + 1024 + h * 64 + ch * 8,
                   sKV + kidx * 72 + ch * 8);
    }
    wait_async0();
    __syncthreads();

    // ---- 2. S = Q K^T, this wave's 16 query rows ----
    const int q0 = qblk * 64 + wv * 16;
    v8f sacc[16];
#pragma unroll
    for (int nt = 0; nt < 16; nt++)
#pragma unroll
        for (int e = 0; e < 8; e++) sacc[nt][e] = 0.0f;

#pragma unroll
    for (int kk = 0; kk < 2; kk++) {
        FragBF fa;
        const int offA = kk * 32 + ((lane & 16) ? 8 : 0);
        const bf16_t* qp = qkv + (rowBase + q0 + (lane & 15)) * 3072 + h * 64 + offA;
        fa.q[0] = *(const uv4*)qp;
        fa.q[1] = *(const uv4*)(qp + 16);
        const int offB = kk * 32 + ((lane & 16) ? 8 : 0);
#pragma unroll
        for (int nt = 0; nt < 16; nt++) {
            FragBF fb;
            const bf16_t* p = sKV + (nt * 16 + (lane & 15)) * 72 + offB;
            fb.q[0] = *(const uv4*)p;
            fb.q[1] = *(const uv4*)(p + 16);
            sacc[nt] = wmma_bf16(fa.v, fb.v, sacc[nt]);
        }
    }

    // ---- 3. softmax over key dim (rows spread across 16-lane halves) ----
    const float scale = 0.125f;            // 1/sqrt(64)
#pragma unroll
    for (int r = 0; r < 8; r++) {
        float mx = -1e30f;
#pragma unroll
        for (int nt = 0; nt < 16; nt++) {
            float v = sacc[nt][r] * scale; sacc[nt][r] = v; mx = fmaxf(mx, v);
        }
#pragma unroll
        for (int d = 1; d < 16; d <<= 1) mx = fmaxf(mx, __shfl_xor(mx, d, 32));
        float sum = 0.0f;
#pragma unroll
        for (int nt = 0; nt < 16; nt++) {
            float p = expf(sacc[nt][r] - mx); sacc[nt][r] = p; sum += p;
        }
#pragma unroll
        for (int d = 1; d < 16; d <<= 1) sum += __shfl_xor(sum, d, 32);
        float rinv = 1.0f / sum;
        int m = r + ((lane & 16) ? 8 : 0);
        bf16_t* prow = sP + (wv * 16 + m) * 264;
#pragma unroll
        for (int nt = 0; nt < 16; nt++)
            prow[nt * 16 + (lane & 15)] = f2bf(sacc[nt][r] * rinv);
    }
    __syncthreads();                        // all waves done with K in LDS

    // ---- 4. V^T -> LDS (reuse K region) [d][kidx] ----
#pragma unroll
    for (int it = 0; it < 16; it++) {
        int u = tid + it * 128;
        int kidx = u >> 3, ch = u & 7;
        uv4 val = *(const uv4*)(qkv + (rowBase + kidx) * 3072 + 2048 + h * 64 + ch * 8);
        bf16_t tmp[8];
        *(uv4*)tmp = val;
#pragma unroll
        for (int e = 0; e < 8; e++) sKV[(ch * 8 + e) * 264 + kidx] = tmp[e];
    }
    __syncthreads();

    // ---- 5. O = P @ V ----
    v8f oacc[4];
#pragma unroll
    for (int j = 0; j < 4; j++)
#pragma unroll
        for (int e = 0; e < 8; e++) oacc[j][e] = 0.0f;

#pragma unroll
    for (int kk = 0; kk < 8; kk++) {
        FragBF fa;
        const int offA = kk * 32 + ((lane & 16) ? 8 : 0);
        const bf16_t* pp = sP + (wv * 16 + (lane & 15)) * 264 + offA;
        fa.q[0] = *(const uv4*)pp;
        fa.q[1] = *(const uv4*)(pp + 16);
        const int offB = kk * 32 + ((lane & 16) ? 8 : 0);
#pragma unroll
        for (int j = 0; j < 4; j++) {
            FragBF fb;
            const bf16_t* p = sKV + (j * 16 + (lane & 15)) * 264 + offB;
            fb.q[0] = *(const uv4*)p;
            fb.q[1] = *(const uv4*)(p + 16);
            oacc[j] = wmma_bf16(fa.v, fb.v, oacc[j]);
        }
    }

    // ---- 6. write bf16 attention output ----
#pragma unroll
    for (int j = 0; j < 4; j++) {
        int n = h * 64 + j * 16 + (lane & 15);
#pragma unroll
        for (int r = 0; r < 8; r++) {
            size_t row = rowBase + q0 + r + ((lane & 16) ? 8 : 0);
            attnO[row * 1024 + n] = f2bf(oacc[j][r]);
        }
    }
}

// ---------------------------------------------------------------------------
// LayerNorm over D=1024, one block per row, bf16 output.
// ---------------------------------------------------------------------------
__global__ __launch_bounds__(256)
void k_ln(const float* __restrict__ x, const float* __restrict__ g,
          const float* __restrict__ b, bf16_t* __restrict__ out)
{
    __shared__ float red[256];
    const int row = blockIdx.x, tid = threadIdx.x;
    const float* xr = x + (size_t)row * 1024;
    float v[4], s = 0.0f;
#pragma unroll
    for (int i = 0; i < 4; i++) { v[i] = xr[tid + i * 256]; s += v[i]; }
    red[tid] = s; __syncthreads();
    for (int st = 128; st > 0; st >>= 1) { if (tid < st) red[tid] += red[tid + st]; __syncthreads(); }
    float mean = red[0] * (1.0f / 1024.0f);
    __syncthreads();
    float s2 = 0.0f;
#pragma unroll
    for (int i = 0; i < 4; i++) { float d = v[i] - mean; s2 += d * d; }
    red[tid] = s2; __syncthreads();
    for (int st = 128; st > 0; st >>= 1) { if (tid < st) red[tid] += red[tid + st]; __syncthreads(); }
    float rstd = rsqrtf(red[0] * (1.0f / 1024.0f) + 1e-5f);
#pragma unroll
    for (int i = 0; i < 4; i++) {
        int c = tid + i * 256;
        out[(size_t)row * 1024 + c] = f2bf((v[i] - mean) * rstd * g[c] + b[c]);
    }
}

// ---------------------------------------------------------------------------
// Gating: logits -> softmax(8) -> top-2 renormalized combine weights + entropy.
// One thread per token row; gate_w cached in LDS.
// ---------------------------------------------------------------------------
__global__ __launch_bounds__(256)
void k_gate(const bf16_t* __restrict__ xn, const float* __restrict__ gw,
            const float* __restrict__ gb, float* __restrict__ combine,
            float* __restrict__ entAcc)
{
    __shared__ float sgw[1024 * 8];
    const int tid = threadIdx.x;
    for (int i = tid; i < 8192; i += 256) sgw[i] = gw[i];
    __syncthreads();

    const int row = blockIdx.x * 256 + tid;
    float lg[8];
#pragma unroll
    for (int e = 0; e < 8; e++) lg[e] = gb[e];
    const bf16_t* xr = xn + (size_t)row * 1024;
    for (int d = 0; d < 1024; d++) {
        float xv = bf2f(xr[d]);
#pragma unroll
        for (int e = 0; e < 8; e++) lg[e] += xv * sgw[d * 8 + e];
    }
    float mx = lg[0];
#pragma unroll
    for (int e = 1; e < 8; e++) mx = fmaxf(mx, lg[e]);
    float pr[8], sum = 0.0f;
#pragma unroll
    for (int e = 0; e < 8; e++) { pr[e] = expf(lg[e] - mx); sum += pr[e]; }
    float rinv = 1.0f / sum;
    float ent = 0.0f;
#pragma unroll
    for (int e = 0; e < 8; e++) { pr[e] *= rinv; ent -= pr[e] * logf(pr[e] + 1e-10f); }

    int i1 = 0;
#pragma unroll
    for (int e = 1; e < 8; e++) if (pr[e] > pr[i1]) i1 = e;
    int i2 = (i1 == 0) ? 1 : 0;
#pragma unroll
    for (int e = 0; e < 8; e++) if (e != i1 && pr[e] > pr[i2]) i2 = e;
    float wsum = pr[i1] + pr[i2];
    float w1 = pr[i1] / wsum, w2 = pr[i2] / wsum;
#pragma unroll
    for (int e = 0; e < 8; e++)
        combine[(size_t)row * 8 + e] = (e == i1) ? w1 : ((e == i2) ? w2 : 0.0f);

    atomicAdd(entAcc, ent);
}

// ---------------------------------------------------------------------------
// Small utility kernels.
// ---------------------------------------------------------------------------
__global__ void k_cvtT(const float* __restrict__ src, bf16_t* __restrict__ dst,
                       int K, int N) {   // dst[n*K+k] = bf16(src[k*N+n])
    long idx = (long)blockIdx.x * 256 + threadIdx.x;
    if (idx < (long)K * N) {
        int n = (int)(idx / K), k = (int)(idx % K);
        dst[idx] = f2bf(src[(size_t)k * N + n]);
    }
}
__global__ void k_cvtSlice(const float* __restrict__ src, bf16_t* __restrict__ dst,
                           long count, int stride, int off) {
    long idx = (long)blockIdx.x * 256 + threadIdx.x;
    if (idx < count) dst[idx] = f2bf(src[idx * stride + off]);
}
__global__ void k_copy(const float* __restrict__ src, float* __restrict__ dst, long n) {
    long idx = (long)blockIdx.x * 256 + threadIdx.x;
    if (idx < n) dst[idx] = src[idx];
}
__global__ void k_zero1(float* p) { if (threadIdx.x == 0) p[0] = 0.0f; }
__global__ void k_final(const float* ent, float* out) {
    if (threadIdx.x == 0) out[0] = 0.1f * ent[0] / 8192.0f;
}

// ---------------------------------------------------------------------------
extern "C" void kernel_launch(void* const* d_in, const int* in_sizes, int n_in,
                              void* d_out, int out_size, void* d_ws, size_t ws_size,
                              hipStream_t stream)
{
    (void)in_sizes; (void)n_in; (void)out_size; (void)ws_size;
    const float* x_in   = (const float*)d_in[0];
    const float* ln1_g  = (const float*)d_in[1];
    const float* ln1_b  = (const float*)d_in[2];
    const float* qkv_w  = (const float*)d_in[3];
    const float* qkv_b  = (const float*)d_in[4];
    const float* ow     = (const float*)d_in[5];
    const float* ob     = (const float*)d_in[6];
    const float* ln2_g  = (const float*)d_in[7];
    const float* ln2_b  = (const float*)d_in[8];
    const float* conv_w = (const float*)d_in[9];
    const float* conv_b = (const float*)d_in[10];
    const float* ln3_g  = (const float*)d_in[11];
    const float* ln3_b  = (const float*)d_in[12];
    const float* gate_w = (const float*)d_in[13];
    const float* gate_b = (const float*)d_in[14];
    const float* exp_w  = (const float*)d_in[15];
    const float* exp_b  = (const float*)d_in[16];
    const float* ff_w1  = (const float*)d_in[17];
    const float* ff_b1  = (const float*)d_in[18];
    const float* ff_w2  = (const float*)d_in[19];
    const float* ff_b2  = (const float*)d_in[20];
    float* out = (float*)d_out;

    const size_t M = 8192;   // B*L
    char* w = (char*)d_ws;
    size_t off = 0;
    auto alloc = [&](size_t bytes) -> char* {
        char* p = w + off;
        off = (off + bytes + 255) & ~(size_t)255;
        return p;
    };
    float*  x1      = (float*)alloc(M * 1024 * 4);
    float*  x2      = (float*)alloc(M * 1024 * 4);
    float*  x3      = (float*)alloc(M * 1024 * 4);
    float*  combine = (float*)alloc(M * 8 * 4);
    float*  entAcc  = (float*)alloc(256);
    bf16_t* xn1     = (bf16_t*)alloc(M * 1024 * 2);
    bf16_t* xn2     = (bf16_t*)alloc(M * 1024 * 2);
    bf16_t* xn3     = (bf16_t*)alloc(M * 1024 * 2);
    bf16_t* qkvb    = (bf16_t*)alloc(M * 3072 * 2);
    bf16_t* attnO   = (bf16_t*)alloc(M * 1024 * 2);
    bf16_t* x3b     = (bf16_t*)alloc(M * 1024 * 2);
    bf16_t* hb      = (bf16_t*)alloc(M * 4096 * 2);
    bf16_t* qkvwT   = (bf16_t*)alloc((size_t)3072 * 1024 * 2);
    bf16_t* owT     = (bf16_t*)alloc((size_t)1024 * 1024 * 2);
    bf16_t* convT   = (bf16_t*)alloc((size_t)3 * 1024 * 1024 * 2);
    bf16_t* expwT   = (bf16_t*)alloc((size_t)8 * 1024 * 1024 * 2);
    bf16_t* ffw1T   = (bf16_t*)alloc((size_t)4096 * 1024 * 2);
    bf16_t* ffw2T   = (bf16_t*)alloc((size_t)4096 * 1024 * 2);

    auto cvtT = [&](const float* s, bf16_t* d, int K, int N) {
        long n = (long)K * N;
        k_cvtT<<<dim3((unsigned)((n + 255) / 256)), dim3(256), 0, stream>>>(s, d, K, N);
    };
    auto gemm = [&](const bf16_t* A, const bf16_t* BT, const float* bias,
                    const float* Cin, const float* resid, const float* rs, int rsStride,
                    float* Cf, bf16_t* Cb, int N, int K, int Lb, int shift, int doGelu) {
        dim3 grid(N / 128, (unsigned)(M / 128));
        k_gemm<<<grid, dim3(256), 0, stream>>>(A, BT, bias, Cin, resid, rs, rsStride,
                                               Cf, Cb, N, K, Lb, shift, doGelu);
    };

    // ---- weight conversion (f32 -> bf16, transposed to [N][K]) ----
    cvtT(qkv_w, qkvwT, 1024, 3072);
    cvtT(ow,    owT,   1024, 1024);
    for (int t = 0; t < 3; t++)   // conv_w[o][i][t] -> convT_t[o*1024+i]
        k_cvtSlice<<<dim3(4096), dim3(256), 0, stream>>>(
            conv_w, convT + (size_t)t * 1024 * 1024, (long)1024 * 1024, 3, t);
    for (int e = 0; e < 8; e++)
        cvtT(exp_w + (size_t)e * 1024 * 1024, expwT + (size_t)e * 1024 * 1024, 1024, 1024);
    cvtT(ff_w1, ffw1T, 1024, 4096);
    cvtT(ff_w2, ffw2T, 4096, 1024);

    // ---- 1. attention branch ----
    k_ln<<<dim3((unsigned)M), dim3(256), 0, stream>>>(x_in, ln1_g, ln1_b, xn1);
    gemm(xn1, qkvwT, qkv_b, nullptr, nullptr, nullptr, 0,
         nullptr, qkvb, 3072, 1024, (int)M, 0, 0);
    k_attn<<<dim3(4, 16, 32), dim3(128), 70656, stream>>>(qkvb, attnO);
    gemm(attnO, owT, ob, nullptr, x_in, nullptr, 0,
         x1, nullptr, 1024, 1024, (int)M, 0, 0);            // x1 = x + attn_out

    // ---- 2. dilated conv branch (3 shifted GEMM taps) ----
    k_ln<<<dim3((unsigned)M), dim3(256), 0, stream>>>(x1, ln2_g, ln2_b, xn2);
    gemm(xn2, convT,                nullptr, nullptr, nullptr, nullptr, 0,
         x2, nullptr, 1024, 1024, 4096, -2, 0);
    gemm(xn2, convT + 1024 * 1024,  nullptr, x2, nullptr, nullptr, 0,
         x2, nullptr, 1024, 1024, 4096,  0, 0);
    gemm(xn2, convT + 2 * 1024 * 1024, conv_b, x2, x1, nullptr, 0,
         x2, nullptr, 1024, 1024, 4096,  2, 0);             // x2 = x1 + conv

    // ---- 3. MoE ----
    k_ln<<<dim3((unsigned)M), dim3(256), 0, stream>>>(x2, ln3_g, ln3_b, xn3);
    k_zero1<<<dim3(1), dim3(64), 0, stream>>>(entAcc);
    k_gate<<<dim3(32), dim3(256), 0, stream>>>(xn3, gate_w, gate_b, combine, entAcc);
    k_copy<<<dim3((unsigned)((M * 1024 + 255) / 256)), dim3(256), 0, stream>>>(
        x2, x3, (long)(M * 1024));
    for (int e = 0; e < 8; e++)
        gemm(xn3, expwT + (size_t)e * 1024 * 1024, exp_b + e * 1024,
             x3, nullptr, combine + e, 8,
             x3, (e == 7) ? x3b : nullptr, 1024, 1024, (int)M, 0, 1);

    // ---- 4. FFN ----
    gemm(x3b, ffw1T, ff_b1, nullptr, nullptr, nullptr, 0,
         nullptr, hb, 4096, 1024, (int)M, 0, 1);
    gemm(hb, ffw2T, ff_b2, nullptr, x3, nullptr, 0,
         out, nullptr, 1024, 4096, (int)M, 0, 0);           // out = x3 + ffn

    // ---- 5. entropy scalar ----
    k_final<<<dim3(1), dim3(64), 0, stream>>>(entAcc, out + M * 1024);
}
